// GShellFlexiCubesGeometry_50886772523023
// MI455X (gfx1250) — compile-verified, matrix operand order checked by reading, not run on hardware
//
#include <hip/hip_runtime.h>

// ---------------------------------------------------------------------------
// GShellFlexiCubes geometry kernel for MI455X (gfx1250, wave32).
// Memory-bound workload (~0.5 GB traffic @ 23.3 TB/s => ~22 us); all effort
// goes into the data-movement path. WMMA (V_WMMA_F32_16X16X4_F32) is used for
// exact-f32 cross-lane reductions of the two scalar reduction trees.
// ---------------------------------------------------------------------------

typedef __attribute__((ext_vector_type(2)))  float    v2f;
typedef __attribute__((ext_vector_type(8)))  float    v8f;
typedef __attribute__((ext_vector_type(16))) _Float16 v16h;

#define NBLK 1024
#define NTHR 256

// ---- wave32 sum via WMMA ---------------------------------------------------
// A (16x4 f32): lane m holds row m (K0 in elem0), lane m+16 holds K2.
// With B = ones, D[m][n] = x_lane(m) + x_lane(m+16).  D row m lives in VGPR m
// (lanes 0-15) / VGPR m-8 (lanes 16-31), so sum of the 8 D regs gives the
// half-wave sum uniformly in each half; one shfl_xor(16) completes the wave.
__device__ __forceinline__ float wave_sum32(float x) {
  v8f d;
#if __has_builtin(__builtin_amdgcn_wmma_f32_16x16x4_f32)
  v2f a; a[0] = x;    a[1] = 0.0f;
  v2f b; b[0] = 1.0f; b[1] = 1.0f;
  v8f c = {};
  d = __builtin_amdgcn_wmma_f32_16x16x4_f32(false, a, false, b, (short)0, c,
                                            false, false);
#else
  // Fallback (codegen-confirmed builtin): hi/lo f16 split keeps ~f32 accuracy
  // because products are exact f16 values accumulated in f32.
  _Float16 hi = (_Float16)x;
  float    r  = x - (float)hi;
  _Float16 lo = (_Float16)r;
  v16h a = {};
  a[0] = hi; a[1] = lo;
  v16h b;
  #pragma unroll
  for (int i = 0; i < 16; ++i) b[i] = (_Float16)1.0f;
  v8f c = {};
  d = __builtin_amdgcn_wmma_f32_16x16x32_f16(false, a, false, b, (short)0, c,
                                             false, false);
#endif
  float s = d[0] + d[1] + d[2] + d[3] + d[4] + d[5] + d[6] + d[7];
  s += __shfl_xor(s, 16, 32);
  return s;  // full-wave sum, uniform across all 32 lanes
}

// Deterministic block sum (valid on thread 0). Safe to call repeatedly with
// the same shared buffer. Requires all threads to participate (EXEC all-ones
// for the WMMA: no early returns in callers).
__device__ __forceinline__ float block_sum(float x, float* sred) {
  float w = wave_sum32(x);
  int lane = threadIdx.x & 31;
  int wid  = threadIdx.x >> 5;
  if (lane == 0) sred[wid] = w;
  __syncthreads();
  float tot = 0.0f;
  if (threadIdx.x == 0) {
    int nw = blockDim.x >> 5;
    for (int i = 0; i < nw; ++i) tot += sred[i];
  }
  __syncthreads();  // allow sred reuse
  return tot;
}

__device__ __forceinline__ float bce_logits(float x, float t) {
  // max(x,0) - x*t + log1p(exp(-|x|))
  return fmaxf(x, 0.0f) - x * t + log1pf(expf(-fabsf(x)));
}

// ---- pass 1: sum of edge vector norms -------------------------------------
__global__ void k_edge_norm(const float* __restrict__ verts,
                            const int2* __restrict__ edges, int E,
                            float* __restrict__ partial) {
  __shared__ float sred[32];
  float acc = 0.0f;
  int stride = gridDim.x * blockDim.x;
  for (int i = blockIdx.x * blockDim.x + threadIdx.x; i < E; i += stride) {
    int2 e = edges[i];
    int b0 = 3 * e.x, b1 = 3 * e.y;
    float dx = verts[b0 + 0] - verts[b1 + 0];
    float dy = verts[b0 + 1] - verts[b1 + 1];
    float dz = verts[b0 + 2] - verts[b1 + 2];
    acc += sqrtf(dx * dx + dy * dy + dz * dz);
  }
  float tot = block_sum(acc, sred);
  if (threadIdx.x == 0) partial[blockIdx.x] = tot;
}

// ---- pass 2: max_disp = (sum/E)/4 ------------------------------------------
__global__ void k_fin_disp(const float* __restrict__ partial, int nblk,
                           float invE, float* __restrict__ ws_md) {
  __shared__ float s[256];
  float a = 0.0f;
  for (int i = threadIdx.x; i < nblk; i += 256) a += partial[i];
  s[threadIdx.x] = a;
  __syncthreads();
  for (int off = 128; off > 0; off >>= 1) {
    if (threadIdx.x < off) s[threadIdx.x] += s[threadIdx.x + off];
    __syncthreads();
  }
  if (threadIdx.x == 0) ws_md[0] = s[0] * invE * 0.25f;
}

// ---- pass 3 (optional): precompute deformed verts into workspace ----------
__global__ void k_vd(const float* __restrict__ verts,
                     const float* __restrict__ deform,
                     const float* __restrict__ ws_md,
                     float* __restrict__ vd, int n3) {
  float md = ws_md[0];
  int stride = gridDim.x * blockDim.x;
  for (int i = blockIdx.x * blockDim.x + threadIdx.x; i < n3; i += stride) {
    float d = fminf(fmaxf(deform[i], -1.0f), 1.0f);
    vd[i] = verts[i] + md * d;
  }
}

__device__ __forceinline__ void load_vd(const float* __restrict__ vdp,
                                        const float* __restrict__ verts,
                                        const float* __restrict__ deform,
                                        float md, int idx,
                                        float& x, float& y, float& z) {
  int b = 3 * idx;
  if (vdp) {
    x = vdp[b + 0]; y = vdp[b + 1]; z = vdp[b + 2];
  } else {
    float dx = fminf(fmaxf(deform[b + 0], -1.0f), 1.0f);
    float dy = fminf(fmaxf(deform[b + 1], -1.0f), 1.0f);
    float dz = fminf(fmaxf(deform[b + 2], -1.0f), 1.0f);
    x = verts[b + 0] + md * dx;
    y = verts[b + 1] + md * dy;
    z = verts[b + 2] + md * dz;
  }
}

// ---- pass 4: per-edge outputs + loss partials ------------------------------
__global__ void k_edges(const float* __restrict__ verts,
                        const float* __restrict__ deform,
                        const float* __restrict__ sdf,
                        const float* __restrict__ msdf,
                        const int2* __restrict__ edges,
                        const float* __restrict__ vdp,  // may be null
                        const float* __restrict__ ws_md,
                        float* __restrict__ out_ev,     // [E][3]
                        float* __restrict__ out_em,     // [E]
                        float* __restrict__ p_bce, float* __restrict__ p_mask,
                        int E) {
  __shared__ float sred[32];
  float md = ws_md[0];
  float acc_b = 0.0f, acc_m = 0.0f;
  int stride = gridDim.x * blockDim.x;
  for (int i = blockIdx.x * blockDim.x + threadIdx.x; i < E; i += stride) {
    int2 e = edges[i];
    float s0 = sdf[e.x];
    float s1 = sdf[e.y];
    // jnp.sign semantics: sign(0) == 0
    int g0 = (s0 > 0.0f) - (s0 < 0.0f);
    int g1 = (s1 > 0.0f) - (s1 < 0.0f);
    bool crossing = (g0 != g1);
    float inv = crossing ? (1.0f / (s1 - s0)) : 0.0f;  // masks output to 0 too

    float v0x, v0y, v0z, v1x, v1y, v1z;
    load_vd(vdp, verts, deform, md, e.x, v0x, v0y, v0z);
    load_vd(vdp, verts, deform, md, e.y, v1x, v1y, v1z);

    float evx = (v0x * s1 - v1x * s0) * inv;
    float evy = (v0y * s1 - v1y * s0) * inv;
    float evz = (v0z * s1 - v1z * s0) * inv;
    float em  = (msdf[e.x] * s1 - msdf[e.y] * s0) * inv;

    // write-once outputs: non-temporal so the 102MB stream skips L2 residency
    __builtin_nontemporal_store(evx, out_ev + 3 * i + 0);
    __builtin_nontemporal_store(evy, out_ev + 3 * i + 1);
    __builtin_nontemporal_store(evz, out_ev + 3 * i + 2);
    __builtin_nontemporal_store(em, out_em + i);

    float t0 = (s1 > 0.0f) ? 1.0f : 0.0f;
    float t1 = (s0 > 0.0f) ? 1.0f : 0.0f;
    float bce = bce_logits(s0, t0) + bce_logits(s1, t1);
    float maskf = crossing ? 1.0f : 0.0f;
    acc_b += maskf * bce;
    acc_m += maskf;
  }
  float tb = block_sum(acc_b, sred);
  float tm = block_sum(acc_m, sred);
  if (threadIdx.x == 0) {
    p_bce[blockIdx.x]  = tb;
    p_mask[blockIdx.x] = tm;
  }
}

// ---- pass 5: reg_loss ------------------------------------------------------
__global__ void k_fin_loss(const float* __restrict__ p_bce,
                           const float* __restrict__ p_mask, int nblk,
                           float* __restrict__ out_reg) {
  __shared__ float sb[256];
  __shared__ float sm[256];
  float a = 0.0f, b = 0.0f;
  for (int i = threadIdx.x; i < nblk; i += 256) {
    a += p_bce[i];
    b += p_mask[i];
  }
  sb[threadIdx.x] = a;
  sm[threadIdx.x] = b;
  __syncthreads();
  for (int off = 128; off > 0; off >>= 1) {
    if (threadIdx.x < off) {
      sb[threadIdx.x] += sb[threadIdx.x + off];
      sm[threadIdx.x] += sm[threadIdx.x + off];
    }
    __syncthreads();
  }
  if (threadIdx.x == 0) out_reg[0] = sb[0] / fmaxf(sm[0], 1.0f);
}

// ---------------------------------------------------------------------------
extern "C" void kernel_launch(void* const* d_in, const int* in_sizes, int n_in,
                              void* d_out, int out_size, void* d_ws,
                              size_t ws_size, hipStream_t stream) {
  const float* verts     = (const float*)d_in[0];
  const float* deform    = (const float*)d_in[1];
  const float* sdf       = (const float*)d_in[2];
  const float* msdf      = (const float*)d_in[3];
  const int2*  all_edges = (const int2*)d_in[4];

  const int n3 = in_sizes[0];      // 3*N
  const int E  = in_sizes[4] / 2;  // number of edges

  float* out    = (float*)d_out;
  float* out_ev = out;                  // E*3
  float* out_em = out + (size_t)3 * E;  // E
  float* reg    = out + (size_t)4 * E;  // 1

  // workspace layout (floats): [0]=max_disp, [16..)=partials, then optional vd
  float* ws     = (float*)d_ws;
  float* ws_md  = ws;
  float* p_norm = ws + 16;
  float* p_bce  = p_norm + NBLK;
  float* p_mask = p_bce + NBLK;
  size_t fixed_bytes = (size_t)(16 + 3 * NBLK) * sizeof(float);
  float* vdp = nullptr;
  if (ws_size >= fixed_bytes + (size_t)n3 * sizeof(float))
    vdp = p_mask + NBLK;  // 3*N floats: deformed verts (saves a verts+deform
                          // re-gather per edge => ~100MB less traffic)

  k_edge_norm<<<NBLK, NTHR, 0, stream>>>(verts, all_edges, E, p_norm);
  k_fin_disp<<<1, 256, 0, stream>>>(p_norm, NBLK, 1.0f / (float)E, ws_md);
  if (vdp)
    k_vd<<<1024, NTHR, 0, stream>>>(verts, deform, ws_md, vdp, n3);
  k_edges<<<NBLK, NTHR, 0, stream>>>(verts, deform, sdf, msdf, all_edges, vdp,
                                     ws_md, out_ev, out_em, p_bce, p_mask, E);
  k_fin_loss<<<1, 256, 0, stream>>>(p_bce, p_mask, NBLK, reg);
}
